// MambaBlock_67886253081031
// MI455X (gfx1250) — matrix operand order repaired and checked by reading, not compiled, not run on hardware
//
#include <hip/hip_runtime.h>
#include <hip/hip_bf16.h>

typedef __bf16 bf16;
typedef __attribute__((ext_vector_type(16))) __bf16 v16bf;
typedef __attribute__((ext_vector_type(8)))  __bf16 v8bf;
typedef __attribute__((ext_vector_type(8)))  float  v8f;

#define NCH  16     // scan chunks
#define CLEN 256    // steps per chunk (NCH*CLEN == L)

// ---------------------------------------------------------------------------
// f32 -> bf16 strided 2D convert
// ---------------------------------------------------------------------------
__global__ void cvt2d_f32_bf16(const float* __restrict__ src, bf16* __restrict__ dst,
                               int rows, int cols, int sstride, int dstride) {
  int idx = blockIdx.x * blockDim.x + threadIdx.x;
  int total = rows * cols;
  if (idx >= total) return;
  int r = idx / cols;
  int c = idx - r * cols;
  dst[(size_t)r * dstride + c] = (bf16)src[(size_t)r * sstride + c];
}

// ---------------------------------------------------------------------------
// bf16 WMMA GEMM:  C(MxN, f32) = A(MxK, bf16 rm) * B(NxK, bf16 rm)^T
// Compile-time tiling, grids sized exactly -> branch-free hot path,
// EXEC all-ones at every WMMA, accumulators pinned.
// EPI==1: C = softplus(acc + bias[n])
// ---------------------------------------------------------------------------
template <int WMW, int WNW, int MT, int NT, int EPI>
__global__ __launch_bounds__(256)
void gemm_bf16_nt(const bf16* __restrict__ A, const bf16* __restrict__ B,
                  float* __restrict__ C, int M, int N, int K,
                  const float* __restrict__ bias) {
  const int lane = threadIdx.x & 31;
  const int wid  = threadIdx.x >> 5;          // 0..7
  const int wm   = wid % WMW;
  const int wn   = wid / WMW;
  const int m0   = blockIdx.x * (WMW * MT * 16) + wm * (MT * 16);
  const int n0   = blockIdx.y * (WNW * NT * 16) + wn * (NT * 16);

  const int lrow = lane & 15;
  const int lhi  = lane >> 4;

  v8f acc[MT][NT];
  #pragma unroll
  for (int tm = 0; tm < MT; ++tm)
    #pragma unroll
    for (int tn = 0; tn < NT; ++tn)
      #pragma unroll
      for (int i = 0; i < 8; ++i) acc[tm][tn][i] = 0.0f;

  const bf16* Abase = A + (size_t)(m0 + lrow) * K + lhi * 16;
  const bf16* Bbase = B + (size_t)(n0 + lrow) * K + lhi * 16;

  for (int k = 0; k < K; k += 32) {
    v16bf a[MT];
    #pragma unroll
    for (int tm = 0; tm < MT; ++tm) {
      const bf16* ap = Abase + (size_t)(tm * 16) * K + k;
      *((v8bf*)&a[tm])     = *(const v8bf*)(ap);
      *((v8bf*)&a[tm] + 1) = *(const v8bf*)(ap + 8);
    }
    #pragma unroll
    for (int tn = 0; tn < NT; ++tn) {
      v16bf b = *(const v16bf*)(Bbase + (size_t)(tn * 16) * K + k);
      #pragma unroll
      for (int tm = 0; tm < MT; ++tm) {
        acc[tm][tn] = __builtin_amdgcn_wmma_f32_16x16x32_bf16(
            false, a[tm], false, b, (short)0, acc[tm][tn], false, false);
      }
    }
  }

  const int nC = lane & 15;
  const int mC = lhi ? 8 : 0;
  #pragma unroll
  for (int tm = 0; tm < MT; ++tm) {
    #pragma unroll
    for (int tn = 0; tn < NT; ++tn) {
      int n = n0 + tn * 16 + nC;
      int m = m0 + tm * 16 + mC;
      #pragma unroll
      for (int r = 0; r < 8; ++r) {
        float v = acc[tm][tn][r];
        if (EPI == 1) {
          v += bias[n];
          v = (v > 20.0f) ? v : log1pf(__expf(v));   // softplus
        }
        C[(size_t)(m + r) * N + n] = v;
      }
    }
  }
}

// ---------------------------------------------------------------------------
// Causal depthwise conv (k=4) + bias + SiLU. Emits f32 (scan) and bf16 (GEMM2).
// ---------------------------------------------------------------------------
__global__ void conv_silu_kernel(const float* __restrict__ xr,
                                 const float* __restrict__ cw,
                                 const float* __restrict__ cb,
                                 float* __restrict__ xs,
                                 bf16* __restrict__ xs_bf,
                                 int L, int DI, int NR) {
  int idx = blockIdx.x * blockDim.x + threadIdx.x;
  if (idx >= L * DI) return;
  int l = idx / DI;
  int d = idx - l * DI;
  float s = cb[d];
  #pragma unroll
  for (int j = 0; j < 4; ++j) {
    int ls = l - 3 + j;
    if (ls >= 0) s += cw[d * 4 + j] * xr[(size_t)ls * NR + d];
  }
  float v = s / (1.0f + __expf(-s));   // silu
  xs[idx]    = v;
  xs_bf[idx] = (bf16)v;
}

// ---------------------------------------------------------------------------
// Chunked selective scan.  h_l = dA_l*h_{l-1} + dBu_l is a diagonal linear
// recurrence -> exact chunk decomposition (16 chunks x 256 steps):
//   pass1: per chunk with h=0, track (P = prod dA, h_end)
//   pass2: h_start(c) = P(c-1)*h_start(c-1) + h_end(c-1)
//   pass3: re-run chunk seeded with h_start, fused output epilogue
// Thread map (pass1/3): 16 lanes per channel (lane = state n).
// ---------------------------------------------------------------------------
__global__ __launch_bounds__(256)
void scan_pass1(const float* __restrict__ delta, const float* __restrict__ dbl,
                const float* __restrict__ xs, const float* __restrict__ A_log,
                float* __restrict__ aprod, float* __restrict__ hend) {
  const int n  = threadIdx.x & 15;
  const int dl = threadIdx.x >> 4;
  const int d  = blockIdx.x * 16 + dl;
  const int c  = blockIdx.y;
  const int DI = 2048, NB = 96;

  const float a = -__expf(A_log[d * 16 + n]);
  float h = 0.0f, P = 1.0f;
  const int l0 = c * CLEN;
  for (int l = l0; l < l0 + CLEN; ++l) {
    float dt = delta[(size_t)l * DI + d];
    float xv = xs[(size_t)l * DI + d];
    float Bv = dbl[(size_t)l * NB + 64 + n];
    float dA = __expf(dt * a);
    h = dA * h + dt * Bv * xv;
    P *= dA;
  }
  int idx = c * (DI * 16) + d * 16 + n;
  aprod[idx] = P;
  hend[idx]  = h;
}

__global__ void scan_pass2(const float* __restrict__ aprod,
                           const float* __restrict__ hend,
                           float* __restrict__ hstart) {
  int i = blockIdx.x * blockDim.x + threadIdx.x;   // (d, n) pair, 32768 total
  if (i >= 2048 * 16) return;
  float h = 0.0f;
  #pragma unroll
  for (int c = 0; c < NCH; ++c) {
    int idx = c * (2048 * 16) + i;
    hstart[idx] = h;
    h = aprod[idx] * h + hend[idx];
  }
}

__global__ __launch_bounds__(256)
void scan_pass3(const float* __restrict__ delta, const float* __restrict__ dbl,
                const float* __restrict__ xs,    const float* __restrict__ xr,
                const float* __restrict__ A_log, const float* __restrict__ Dp,
                const float* __restrict__ hstart, bf16* __restrict__ ybf) {
  const int n  = threadIdx.x & 15;
  const int dl = threadIdx.x >> 4;
  const int d  = blockIdx.x * 16 + dl;
  const int c  = blockIdx.y;
  const int DI = 2048, NR = 4096, NB = 96;

  const float a  = -__expf(A_log[d * 16 + n]);
  const float Dd = Dp[d];
  float h = hstart[c * (DI * 16) + d * 16 + n];

  const int l0 = c * CLEN;
  for (int l = l0; l < l0 + CLEN; ++l) {
    float dt = delta[(size_t)l * DI + d];
    float xv = xs[(size_t)l * DI + d];
    float Bv = dbl[(size_t)l * NB + 64 + n];
    float Cv = dbl[(size_t)l * NB + 80 + n];
    float dA = __expf(dt * a);
    h = dA * h + dt * Bv * xv;
    float y = h * Cv;
    y += __shfl_xor(y, 8, 16);
    y += __shfl_xor(y, 4, 16);
    y += __shfl_xor(y, 2, 16);
    y += __shfl_xor(y, 1, 16);
    if (n == 0) {
      float r = xr[(size_t)l * NR + 2048 + d];
      float g = r / (1.0f + __expf(-r));       // silu(res)
      ybf[(size_t)l * DI + d] = (bf16)((y + xv * Dd) * g);
    }
  }
}

// ---------------------------------------------------------------------------
extern "C" void kernel_launch(void* const* d_in, const int* in_sizes, int n_in,
                              void* d_out, int out_size, void* d_ws, size_t ws_size,
                              hipStream_t stream) {
  const float* x      = (const float*)d_in[0];
  const float* W_in   = (const float*)d_in[1];
  const float* conv_w = (const float*)d_in[2];
  const float* conv_b = (const float*)d_in[3];
  const float* W_x    = (const float*)d_in[4];
  const float* W_dt   = (const float*)d_in[5];
  const float* b_dt   = (const float*)d_in[6];
  const float* A_log  = (const float*)d_in[7];
  const float* Dp     = (const float*)d_in[8];
  const float* W_out  = (const float*)d_in[9];
  float* out = (float*)d_out;

  const int L = 4096, DM = 1024, DI = 2048, NR = 4096, NX = 96, DTR = 64;

  char*  ws  = (char*)d_ws;
  size_t off = 0;
  auto alloc = [&](size_t bytes) -> void* {
    void* p = ws + off;
    off = (off + bytes + 255) & ~(size_t)255;
    return p;
  };
  float* xr     = (float*)alloc((size_t)L * NR * 4);   // 64 MB (xi | res)
  float* xs     = (float*)alloc((size_t)L * DI * 4);   // 32 MB
  float* dblv   = (float*)alloc((size_t)L * NX * 4);   // 1.5 MB (dt_in|B|C)
  float* delta  = (float*)alloc((size_t)L * DI * 4);   // 32 MB
  float* aprod  = (float*)alloc((size_t)NCH * DI * 16 * 4);  // 2 MB
  float* hend   = (float*)alloc((size_t)NCH * DI * 16 * 4);  // 2 MB
  float* hstart = (float*)alloc((size_t)NCH * DI * 16 * 4);  // 2 MB
  bf16*  x_bf   = (bf16*)alloc((size_t)L * DM * 2);
  bf16*  Win_bf = (bf16*)alloc((size_t)NR * DM * 2);
  bf16*  Wx_bf  = (bf16*)alloc((size_t)NX * DI * 2);
  bf16*  Wdt_bf = (bf16*)alloc((size_t)DI * DTR * 2);
  bf16*  Wout_bf= (bf16*)alloc((size_t)DM * DI * 2);
  bf16*  xs_bf  = (bf16*)alloc((size_t)L * DI * 2);
  bf16*  dt_bf  = (bf16*)alloc((size_t)L * DTR * 2);
  bf16*  y_bf   = (bf16*)alloc((size_t)L * DI * 2);

  auto cvt = [&](const float* s, bf16* dp, int rows, int cols, int sstr, int dstr) {
    int total = rows * cols;
    cvt2d_f32_bf16<<<(total + 255) / 256, 256, 0, stream>>>(s, dp, rows, cols, sstr, dstr);
  };
  cvt(x,     x_bf,   L,  DM,  DM,  DM);
  cvt(W_in,  Win_bf, NR, DM,  DM,  DM);
  cvt(W_x,   Wx_bf,  NX, DI,  DI,  DI);
  cvt(W_dt,  Wdt_bf, DI, DTR, DTR, DTR);
  cvt(W_out, Wout_bf,DM, DI,  DI,  DI);

  dim3 blk(256);

  // GEMM1: xr = x @ W_in.T            (4096 x 4096 x 1024), tile 64x256
  gemm_bf16_nt<2, 4, 2, 4, 0><<<dim3(L / 64, NR / 256), blk, 0, stream>>>(
      x_bf, Win_bf, xr, L, NR, DM, nullptr);

  // conv + silu -> xs (f32 + bf16)
  conv_silu_kernel<<<((L * DI) + 255) / 256, 256, 0, stream>>>(
      xr, conv_w, conv_b, xs, xs_bf, L, DI, NR);

  // GEMM2: dbl = xs @ W_x.T           (4096 x 96 x 2048), tile 128x96
  gemm_bf16_nt<8, 1, 1, 6, 0><<<dim3(L / 128, 1), blk, 0, stream>>>(
      xs_bf, Wx_bf, dblv, L, NX, DI, nullptr);

  // dt_in (cols 0..63 of dbl) -> bf16
  cvt(dblv, dt_bf, L, DTR, NX, DTR);

  // GEMM3: delta = softplus(dt_in @ W_dt.T + b_dt)   (4096 x 2048 x 64)
  gemm_bf16_nt<2, 4, 2, 4, 1><<<dim3(L / 64, DI / 256), blk, 0, stream>>>(
      dt_bf, Wdt_bf, delta, L, DI, DTR, b_dt);

  // chunked selective scan (exact): pass1 -> pass2 -> pass3 (+gating -> y_bf)
  scan_pass1<<<dim3(DI / 16, NCH), blk, 0, stream>>>(
      delta, dblv, xs, A_log, aprod, hend);
  scan_pass2<<<(DI * 16 + 255) / 256, 256, 0, stream>>>(aprod, hend, hstart);
  scan_pass3<<<dim3(DI / 16, NCH), blk, 0, stream>>>(
      delta, dblv, xs, xr, A_log, Dp, hstart, y_bf);

  // GEMM4: out = y @ W_out.T          (4096 x 1024 x 2048)
  gemm_bf16_nt<2, 4, 2, 4, 0><<<dim3(L / 64, DM / 256), blk, 0, stream>>>(
      y_bf, Wout_bf, out, L, DM, DI, nullptr);
}
